// GroupMultiHeadAttention_36919538876722
// MI455X (gfx1250) — matrix operand order repaired and checked by reading, not compiled, not run on hardware
//
#include <hip/hip_runtime.h>
#include <hip/hip_bf16.h>
#include <math.h>
#include <stdint.h>

typedef _Float16 half_t;
typedef __attribute__((ext_vector_type(16))) _Float16 v16h;
typedef __attribute__((ext_vector_type(8)))  _Float16 v8h;
typedef __attribute__((ext_vector_type(4)))  _Float16 v4h;
typedef __attribute__((ext_vector_type(8)))  float    v8f;
typedef int b128_t __attribute__((vector_size(16)));   // matches builtin param type

#define L_SEQ   2048
#define D_MODEL 2048
#define NH      32
#define NG      8
#define HDIM    64
#define QLD     (NH * HDIM)   // 2048
#define KLD     (NG * HDIM)   // 512

// ---------------------------------------------------------------------------
// Async Global -> LDS copy (CDNA5 GLOBAL_LOAD_ASYNC_TO_LDS_B128, ASYNCcnt).
// Probe-confirmed signature: (int4 AS1*, int4 AS3*, imm offset, imm cpol).
#if defined(__has_builtin)
#  if __has_builtin(__builtin_amdgcn_global_load_async_to_lds_b128)
#    define ATHENA_HAVE_ASYNC_LDS 1
#  endif
#endif
#ifndef ATHENA_HAVE_ASYNC_LDS
#  define ATHENA_HAVE_ASYNC_LDS 0
#endif

__device__ __forceinline__ void copy16_to_lds(const half_t* __restrict__ g, half_t* l) {
#if ATHENA_HAVE_ASYNC_LDS
  __builtin_amdgcn_global_load_async_to_lds_b128(
      (__attribute__((address_space(1))) b128_t*)(uintptr_t)g,
      (__attribute__((address_space(3))) b128_t*)(uint32_t)(uintptr_t)l,
      /*offset=*/0, /*cpol=*/0);
#else
  *(v8h*)l = *(const v8h*)g;
#endif
}

__device__ __forceinline__ void lds_copy_fence() {
#if ATHENA_HAVE_ASYNC_LDS
#  if defined(__has_builtin) && __has_builtin(__builtin_amdgcn_s_wait_asynccnt)
  __builtin_amdgcn_s_wait_asynccnt(0);
#  else
  asm volatile("s_wait_asynccnt 0x0" ::: "memory");
#  endif
#endif
  // Synchronous fallback needs nothing: __syncthreads() waits DScnt.
}

// ---------------------------------------------------------------------------
// WMMA fragment loaders (wave32), contiguous k = 16*(lane/16) + s (not the ISA table order); A and B match.
// A (16x32 f16, MxK): lanes 0-15 -> M=lane, K 0..15;
//                     lanes 16-31 -> M=lane-16, K 16..31.
__device__ __forceinline__ v16h load_afrag(const half_t* __restrict__ base, int ld) {
  const int lane = threadIdx.x & 31;
  const int m  = lane & 15;
  const int hi = lane >> 4;
  const half_t* r = base + (size_t)m * ld + hi * 16;
  v16h a;
#pragma unroll
  for (int i = 0; i < 8; ++i) a[i] = r[i];
#pragma unroll
  for (int i = 0; i < 8; ++i) a[8 + i] = r[8 + i];
  return a;
}

// B (32x16 f16, KxN) where source holds B[k][n] = base[n*ld + k]
// (N-major array of K-length rows, e.g. weight matrices W[N][K] or K tiles).
__device__ __forceinline__ v16h load_bfrag_nmajor(const half_t* __restrict__ base, int ld) {
  const int lane = threadIdx.x & 31;
  const int n  = lane & 15;
  const int hi = lane >> 4;
  const half_t* r = base + (size_t)n * ld + hi * 16;
  v16h b;
#pragma unroll
  for (int i = 0; i < 16; ++i) b[i] = r[i];
  return b;
}

// B (32x16 f16, KxN) where source holds B[k][n] = base[k*ld + n] (K-major, e.g. V tile).
__device__ __forceinline__ v16h load_bfrag_kmajor(const half_t* __restrict__ base, int ld) {
  const int lane = threadIdx.x & 31;
  const int n  = lane & 15;
  const int hi = lane >> 4;
  const half_t* c = base + (size_t)(hi * 16) * ld + n;
  v16h b;
#pragma unroll
  for (int i = 0; i < 16; ++i) b[i] = c[(size_t)i * ld];
  return b;
}

// ---------------------------------------------------------------------------
// fp32 -> f16 conversion (vectorized by 4)
__global__ void cvt_kernel(const float* __restrict__ in, half_t* __restrict__ out, int n) {
  int i = (blockIdx.x * blockDim.x + threadIdx.x) * 4;
  if (i >= n) return;
  float4 v = *(const float4*)(in + i);
  v4h o;
  o[0] = (half_t)v.x; o[1] = (half_t)v.y; o[2] = (half_t)v.z; o[3] = (half_t)v.w;
  *(v4h*)(out + i) = o;
}

// ---------------------------------------------------------------------------
// Generic C[MxN] = A[MxK] * W[NxK]^T via v_wmma_f32_16x16x32_f16.
// Block tile 128x128x64, 8 waves (32x64 each), double-buffered async LDS staging.
// Dynamic LDS: 2 * (128*72 + 128*72) halfs = 73728 bytes.
template <typename OutT>
__global__ __launch_bounds__(256) void wmma_gemm_nt(
    const half_t* __restrict__ A, const half_t* __restrict__ W,
    OutT* __restrict__ C, int M, int N, int K) {
  constexpr int BM = 128, BN = 128, BK = 64;
  constexpr int LDA = BK + 8;  // 72 halfs: pad so A-frag rows hit distinct banks

  extern __shared__ __align__(16) char smem_raw[];
  half_t* sAbuf[2];
  half_t* sWbuf[2];
  sAbuf[0] = (half_t*)smem_raw;
  sAbuf[1] = sAbuf[0] + BM * LDA;
  sWbuf[0] = sAbuf[1] + BM * LDA;
  sWbuf[1] = sWbuf[0] + BN * LDA;

  const int m0 = blockIdx.y * BM;
  const int n0 = blockIdx.x * BN;
  const int tid  = threadIdx.x;
  const int wave = tid >> 5;
  const int wm = wave >> 1;  // 0..3 : 32-row slab
  const int wn = wave & 1;   // 0..1 : 64-col slab

  auto stage = [&](int buf, int k0) {
#pragma unroll
    for (int t = 0; t < 4; ++t) {
      int c  = tid + 256 * t;
      int r  = c >> 3;
      int cp = (c & 7) * 8;
      copy16_to_lds(A + (size_t)(m0 + r) * K + k0 + cp, &sAbuf[buf][r * LDA + cp]);
      copy16_to_lds(W + (size_t)(n0 + r) * K + k0 + cp, &sWbuf[buf][r * LDA + cp]);
    }
  };

  v8f acc[2][4];
#pragma unroll
  for (int i = 0; i < 2; ++i)
#pragma unroll
    for (int j = 0; j < 4; ++j)
#pragma unroll
      for (int r = 0; r < 8; ++r) acc[i][j][r] = 0.0f;

  const int nk = K / BK;
  stage(0, 0);
  for (int kt = 0; kt < nk; ++kt) {
    lds_copy_fence();      // my async copies into buf[kt&1] (and earlier) are done
    __syncthreads();       // everyone's copies visible; prior reads of other buf done
    if (kt + 1 < nk) stage((kt + 1) & 1, (kt + 1) * BK);  // overlap with compute

    const half_t* sA = sAbuf[kt & 1];
    const half_t* sW = sWbuf[kt & 1];

    v16h af[2][2], bf[4][2];
#pragma unroll
    for (int i = 0; i < 2; ++i)
#pragma unroll
      for (int ks = 0; ks < 2; ++ks)
        af[i][ks] = load_afrag(&sA[(wm * 32 + i * 16) * LDA + ks * 32], LDA);
#pragma unroll
    for (int j = 0; j < 4; ++j)
#pragma unroll
      for (int ks = 0; ks < 2; ++ks)
        bf[j][ks] = load_bfrag_nmajor(&sW[(wn * 64 + j * 16) * LDA + ks * 32], LDA);

#pragma unroll
    for (int i = 0; i < 2; ++i)
#pragma unroll
      for (int j = 0; j < 4; ++j)
#pragma unroll
        for (int ks = 0; ks < 2; ++ks)
          acc[i][j] = __builtin_amdgcn_wmma_f32_16x16x32_f16(
              false, af[i][ks], false, bf[j][ks], (short)0, acc[i][j], false, false);
    // No trailing barrier: the next iteration's top barrier separates this
    // iteration's LDS reads from the stage() that overwrites this buffer.
  }

  const int lane = tid & 31;
  const int n_in = lane & 15;
  const int hi   = lane >> 4;
#pragma unroll
  for (int i = 0; i < 2; ++i)
#pragma unroll
    for (int j = 0; j < 4; ++j)
#pragma unroll
      for (int r = 0; r < 8; ++r) {
        int mm = m0 + wm * 32 + i * 16 + hi * 8 + r;
        int nn = n0 + wn * 64 + j * 16 + n_in;
        C[(size_t)mm * N + nn] = (OutT)acc[i][j][r];
      }
}

// ---------------------------------------------------------------------------
// In-place RoPE on an [L][nheads*64] f16 buffer. One thread per (pos, head, d<32) pair.
__global__ void rope_kernel(half_t* __restrict__ data, const float* __restrict__ cosb,
                            const float* __restrict__ sinb, int nheads) {
  int idx = blockIdx.x * blockDim.x + threadIdx.x;
  int d = idx & 31;
  int h = (idx >> 5) % nheads;
  int i = idx / (nheads * 32);
  if (i >= L_SEQ) return;
  half_t* p = data + (size_t)i * nheads * HDIM + h * HDIM;
  float x1 = (float)p[d], x2 = (float)p[d + 32];
  float c1 = cosb[i * HDIM + d],      s1 = sinb[i * HDIM + d];
  float c2 = cosb[i * HDIM + d + 32], s2 = sinb[i * HDIM + d + 32];
  p[d]      = (half_t)(x1 * c1 - x2 * s1);
  p[d + 32] = (half_t)(x2 * c2 + x1 * s2);
}

// ---------------------------------------------------------------------------
// Flash attention: block = (head, 64 query rows), 4 waves, each wave owns 16 rows.
// K/V tiles double-buffered with async LDS staging; causal mask analytic.
__global__ __launch_bounds__(128) void flash_attn_kernel(
    const half_t* __restrict__ qh, const half_t* __restrict__ kh,
    const half_t* __restrict__ vh, half_t* __restrict__ oh) {
  constexpr int LDK = HDIM + 8;  // 72
  __shared__ __align__(16) half_t sK[2][64 * LDK];
  __shared__ __align__(16) half_t sV[2][64 * LDK];
  __shared__ __align__(16) half_t sP[4][16 * LDK];

  const int h   = blockIdx.y;
  const int g   = h >> 2;            // S = H/G = 4 query heads per kv head
  const int qb  = blockIdx.x;
  const int tid = threadIdx.x;
  const int wave = tid >> 5;
  const int lane = tid & 31;
  const int n_in = lane & 15;
  const int hi   = lane >> 4;
  const int qrow0 = qb * 64 + wave * 16;

  auto stage = [&](int buf, int jb) {
    const int j0 = jb * 64;
#pragma unroll
    for (int t = 0; t < 4; ++t) {
      int c  = tid + 128 * t;
      int r  = c >> 3;
      int cp = (c & 7) * 8;
      copy16_to_lds(kh + (size_t)(j0 + r) * KLD + g * HDIM + cp, &sK[buf][r * LDK + cp]);
      copy16_to_lds(vh + (size_t)(j0 + r) * KLD + g * HDIM + cp, &sV[buf][r * LDK + cp]);
    }
  };

  v16h aQ[2];
#pragma unroll
  for (int ks = 0; ks < 2; ++ks)
    aQ[ks] = load_afrag(qh + (size_t)qrow0 * QLD + h * HDIM + ks * 32, QLD);

  float m_i[8], l_i[8];
  v8f Oacc[4];
#pragma unroll
  for (int r = 0; r < 8; ++r) { m_i[r] = -__builtin_inff(); l_i[r] = 0.f; }
#pragma unroll
  for (int d = 0; d < 4; ++d)
#pragma unroll
    for (int r = 0; r < 8; ++r) Oacc[d][r] = 0.f;

  stage(0, 0);
  for (int jb = 0; jb <= qb; ++jb) {
    lds_copy_fence();
    __syncthreads();
    if (jb < qb) stage((jb + 1) & 1, jb + 1);  // prefetch next K/V block

    const half_t* cK = &sK[jb & 1][0];
    const half_t* cV = &sV[jb & 1][0];
    const int j0 = jb * 64;

    // S = Q K^T (16x64 per wave)
    v8f S[4];
#pragma unroll
    for (int j = 0; j < 4; ++j) {
#pragma unroll
      for (int r = 0; r < 8; ++r) S[j][r] = 0.f;
#pragma unroll
      for (int ks = 0; ks < 2; ++ks) {
        v16h bk = load_bfrag_nmajor(&cK[(j * 16) * LDK + ks * 32], LDK);
        S[j] = __builtin_amdgcn_wmma_f32_16x16x32_f16(false, aQ[ks], false, bk,
                                                      (short)0, S[j], false, false);
      }
    }
    const bool diag = (jb == qb);
#pragma unroll
    for (int j = 0; j < 4; ++j)
#pragma unroll
      for (int r = 0; r < 8; ++r) {
        float s = S[j][r] * 0.125f;  // 1/sqrt(HD)
        if (diag) {
          int col = j0 + j * 16 + n_in;
          int row = qrow0 + hi * 8 + r;
          if (col > row) s = -__builtin_inff();
        }
        S[j][r] = s;
      }

    // online softmax; xor masks 1..8 reduce across the 16-lane group owning one N set
#pragma unroll
    for (int r = 0; r < 8; ++r) {
      float mrow = fmaxf(fmaxf(S[0][r], S[1][r]), fmaxf(S[2][r], S[3][r]));
#pragma unroll
      for (int off = 8; off >= 1; off >>= 1)
        mrow = fmaxf(mrow, __shfl_xor(mrow, off, 32));
      float mnew = fmaxf(m_i[r], mrow);
      float corr = __expf(m_i[r] - mnew);
      float lrow = 0.f;
#pragma unroll
      for (int j = 0; j < 4; ++j) {
        float p = __expf(S[j][r] - mnew);
        S[j][r] = p;
        lrow += p;
      }
#pragma unroll
      for (int off = 8; off >= 1; off >>= 1)
        lrow += __shfl_xor(lrow, off, 32);
      l_i[r] = l_i[r] * corr + lrow;
      m_i[r] = mnew;
#pragma unroll
      for (int d = 0; d < 4; ++d) Oacc[d][r] *= corr;
    }

    // C-layout -> A-layout via wave-private LDS (same-wave DS ops are in-order),
    // then O += P @ V.
    half_t* sPw = &sP[wave][0];
#pragma unroll
    for (int j = 0; j < 4; ++j)
#pragma unroll
      for (int r = 0; r < 8; ++r)
        sPw[(hi * 8 + r) * LDK + j * 16 + n_in] = (half_t)S[j][r];

    v16h aP[2];
#pragma unroll
    for (int ks = 0; ks < 2; ++ks) aP[ks] = load_afrag(sPw + ks * 32, LDK);
#pragma unroll
    for (int d = 0; d < 4; ++d)
#pragma unroll
      for (int ks = 0; ks < 2; ++ks) {
        v16h bv = load_bfrag_kmajor(&cV[(ks * 32) * LDK + d * 16], LDK);
        Oacc[d] = __builtin_amdgcn_wmma_f32_16x16x32_f16(false, aP[ks], false, bv,
                                                         (short)0, Oacc[d], false, false);
      }
    // No trailing barrier: next top barrier separates buffer reuse.
  }

#pragma unroll
  for (int d = 0; d < 4; ++d)
#pragma unroll
    for (int r = 0; r < 8; ++r) {
      int row = qrow0 + hi * 8 + r;
      float v = Oacc[d][r] / l_i[r];
      oh[(size_t)row * QLD + h * HDIM + d * 16 + n_in] = (half_t)v;
    }
}

// ---------------------------------------------------------------------------
extern "C" void kernel_launch(void* const* d_in, const int* in_sizes, int n_in,
                              void* d_out, int out_size, void* d_ws, size_t ws_size,
                              hipStream_t stream) {
  (void)in_sizes; (void)n_in; (void)out_size; (void)ws_size;
  const float* x    = (const float*)d_in[0];
  // d_in[1] is the boolean causal mask: applied analytically, never read.
  const float* cosb = (const float*)d_in[2];
  const float* sinb = (const float*)d_in[3];
  const float* Wq   = (const float*)d_in[4];
  const float* Wk   = (const float*)d_in[5];
  const float* Wv   = (const float*)d_in[6];
  const float* Wo   = (const float*)d_in[7];
  float* out = (float*)d_out;

  char* ws = (char*)d_ws;
  const size_t MB = 1u << 20;
  half_t* xh  = (half_t*)(ws + 0 * MB);    // 8 MB  x in f16
  half_t* Wqh = (half_t*)(ws + 8 * MB);    // 8 MB
  half_t* Wkh = (half_t*)(ws + 16 * MB);   // 2 MB
  half_t* Wvh = (half_t*)(ws + 18 * MB);   // 2 MB
  half_t* Woh = (half_t*)(ws + 20 * MB);   // 8 MB
  half_t* qhp = (half_t*)(ws + 28 * MB);   // 8 MB  [L][H*64]
  half_t* khp = (half_t*)(ws + 36 * MB);   // 2 MB  [L][G*64]
  half_t* vhp = (half_t*)(ws + 38 * MB);   // 2 MB  [L][G*64]
  half_t* ohp = (half_t*)(ws + 40 * MB);   // 8 MB  attention output [L][H*64]

  auto cvt = [&](const float* src, half_t* dst, int n) {
    cvt_kernel<<<dim3((n / 4 + 255) / 256), dim3(256), 0, stream>>>(src, dst, n);
  };
  cvt(x,  xh,  L_SEQ * D_MODEL);
  cvt(Wq, Wqh, D_MODEL * D_MODEL);
  cvt(Wk, Wkh, KLD * D_MODEL);
  cvt(Wv, Wvh, KLD * D_MODEL);
  cvt(Wo, Woh, D_MODEL * D_MODEL);

  constexpr size_t GEMM_LDS = (size_t)2 * 2 * 128 * 72 * sizeof(half_t);  // 73728 B

  wmma_gemm_nt<half_t><<<dim3(D_MODEL / 128, L_SEQ / 128), 256, GEMM_LDS, stream>>>(
      xh, Wqh, qhp, L_SEQ, D_MODEL, D_MODEL);
  wmma_gemm_nt<half_t><<<dim3(KLD / 128, L_SEQ / 128), 256, GEMM_LDS, stream>>>(
      xh, Wkh, khp, L_SEQ, KLD, D_MODEL);
  wmma_gemm_nt<half_t><<<dim3(KLD / 128, L_SEQ / 128), 256, GEMM_LDS, stream>>>(
      xh, Wvh, vhp, L_SEQ, KLD, D_MODEL);

  rope_kernel<<<dim3(L_SEQ * NH * 32 / 256), 256, 0, stream>>>(qhp, cosb, sinb, NH);
  rope_kernel<<<dim3(L_SEQ * NG * 32 / 256), 256, 0, stream>>>(khp, cosb, sinb, NG);

  flash_attn_kernel<<<dim3(L_SEQ / 64, NH), 128, 0, stream>>>(qhp, khp, vhp, ohp);

  wmma_gemm_nt<float><<<dim3(D_MODEL / 128, L_SEQ / 128), 256, GEMM_LDS, stream>>>(
      ohp, Woh, out, L_SEQ, D_MODEL, D_MODEL);
}
